// TT_56788057588463
// MI455X (gfx1250) — matrix operand order repaired and checked by reading, not compiled, not run on hardware
//
#include <hip/hip_runtime.h>
#include <hip/hip_bf16.h>

// ---------------------------------------------------------------------------
// TT layer forward on gfx1250:
//   G = reshape( reshape(g0(256x64) @ g1(64x16384), 65536x64) @ g2(64x256),
//                4096, 4096)
//   out = X(8192x4096) @ G + bias
// Main GEMM: bf16 WMMA (f32 acc), 128x128 workgroup macro-tiles, K-step 64,
// operands double-buffered in LDS via GLOBAL_LOAD_ASYNC_TO_LDS_B128.
// ---------------------------------------------------------------------------

typedef __attribute__((ext_vector_type(16))) __bf16 v16bf;
typedef __attribute__((ext_vector_type(8)))  float  v8f;

union Frag16 {
    uint4  u[2];   // two 16-byte halves
    v16bf  v;      // 16 bf16 elements (8 VGPRs)
};

__device__ __forceinline__ unsigned short f32_to_bf16_rne(float f) {
    unsigned int u = __builtin_bit_cast(unsigned int, f);
    u += 0x7FFFu + ((u >> 16) & 1u);          // round to nearest even
    return (unsigned short)(u >> 16);
}

// ---------------------------------------------------------------------------
// Kernel 1: A1[p][q] = sum_a g0[p][a] * g1[a][q]   (256x16384 fp32)
// ---------------------------------------------------------------------------
__global__ __launch_bounds__(256) void tt_build_a1(const float* __restrict__ g0,
                                                   const float* __restrict__ g1,
                                                   float* __restrict__ A1) {
    const int t  = blockIdx.x * 256 + threadIdx.x;   // 1,048,576 threads
    const int q  = t & 16383;
    const int p0 = (t >> 14) * 4;                    // 64 groups of 4 rows

    float acc0 = 0.f, acc1 = 0.f, acc2 = 0.f, acc3 = 0.f;
    #pragma unroll 8
    for (int a = 0; a < 64; ++a) {
        const float g1v = g1[(size_t)a * 16384 + q]; // coalesced across lanes
        acc0 += g0[(p0 + 0) * 64 + a] * g1v;         // uniform -> scalar loads
        acc1 += g0[(p0 + 1) * 64 + a] * g1v;
        acc2 += g0[(p0 + 2) * 64 + a] * g1v;
        acc3 += g0[(p0 + 3) * 64 + a] * g1v;
    }
    A1[(size_t)(p0 + 0) * 16384 + q] = acc0;
    A1[(size_t)(p0 + 1) * 16384 + q] = acc1;
    A1[(size_t)(p0 + 2) * 16384 + q] = acc2;
    A1[(size_t)(p0 + 3) * 16384 + q] = acc3;
}

// ---------------------------------------------------------------------------
// Kernel 2: Gt[c][r] = bf16( sum_b A1[(r*16+l)*64+b] * g2[b*256+mn] )
//   c = l*256 + mn. G stored TRANSPOSED (N-major) so WMMA B loads are
//   contiguous in K. LDS-tiled.
// ---------------------------------------------------------------------------
__global__ __launch_bounds__(256) void tt_build_gt(const float* __restrict__ A1,
                                                   const float* __restrict__ g2,
                                                   unsigned short* __restrict__ Gt) {
    __shared__ float a1s[16 * 64];     // 4 KB
    __shared__ float g2s[64 * 128];    // 32 KB

    const int bid    = blockIdx.x;     // 8192 blocks
    const int mnHalf = bid & 1;
    const int l      = (bid >> 1) & 15;
    const int rg     = bid >> 5;       // 0..255
    const int tid    = threadIdx.x;

    {
        const int idx = tid * 4;             // 0..1020
        const int rr  = idx >> 6;
        const int b   = idx & 63;
        const float4 v =
            *(const float4*)&A1[(size_t)(((rg * 16 + rr) * 16 + l)) * 64 + b];
        *(float4*)&a1s[idx] = v;
    }
    #pragma unroll
    for (int i = 0; i < 32; ++i) {
        const int li  = tid + i * 256;
        const int b   = li >> 7;
        const int col = li & 127;
        g2s[li] = g2[(size_t)b * 256 + mnHalf * 128 + col];
    }
    __syncthreads();

    const int col   = tid & 127;
    const int rbase = (tid >> 7) * 8;    // 0 or 8

    float acc[8] = {0.f, 0.f, 0.f, 0.f, 0.f, 0.f, 0.f, 0.f};
    #pragma unroll 4
    for (int b = 0; b < 64; ++b) {
        const float g = g2s[b * 128 + col];
        #pragma unroll
        for (int j = 0; j < 8; ++j)
            acc[j] += a1s[(rbase + j) * 64 + b] * g;
    }

    unsigned short pk[8];
    #pragma unroll
    for (int j = 0; j < 8; ++j) pk[j] = f32_to_bf16_rne(acc[j]);
    const int c  = l * 256 + mnHalf * 128 + col;
    const int r0 = rg * 16 + rbase;
    *(uint4*)&Gt[(size_t)c * 4096 + r0] = *(const uint4*)pk;
}

// ---------------------------------------------------------------------------
// Kernel 3: convert X (fp32) -> Xb (bf16), 8 elements per thread.
// ---------------------------------------------------------------------------
__global__ __launch_bounds__(256) void tt_cvt_x(const float* __restrict__ X,
                                                unsigned short* __restrict__ Xb) {
    const size_t base = ((size_t)blockIdx.x * 256 + threadIdx.x) * 8;
    const float4 f0 = *(const float4*)&X[base];
    const float4 f1 = *(const float4*)&X[base + 4];
    unsigned short pk[8];
    pk[0] = f32_to_bf16_rne(f0.x); pk[1] = f32_to_bf16_rne(f0.y);
    pk[2] = f32_to_bf16_rne(f0.z); pk[3] = f32_to_bf16_rne(f0.w);
    pk[4] = f32_to_bf16_rne(f1.x); pk[5] = f32_to_bf16_rne(f1.y);
    pk[6] = f32_to_bf16_rne(f1.z); pk[7] = f32_to_bf16_rne(f1.w);
    *(uint4*)&Xb[base] = *(const uint4*)pk;
}

// ---------------------------------------------------------------------------
// Kernel 4: out(8192x4096) = Xb @ G + bias via WMMA bf16.
//   Workgroup = 256 threads (8 waves), macro-tile 128(M) x 128(N), K-step 64.
//   A panel 128x64 bf16 + B panel (N-major) 128x64 bf16 staged in LDS with
//   global_load_async_to_lds_b128, double-buffered (72 KB LDS total).
//   LDS rows padded to 72 elems (144 B = 36 banks, gcd(36,64)=4 -> conflict-
//   free b128 reads). Waves 4(M) x 2(N); each computes 32x64 = 2x4 WMMA tiles
//   per 32-wide K sub-step (16 WMMAs per LDS stage).
//   Steady-state loop is branch-free: last iteration peeled.
// ---------------------------------------------------------------------------
#define TT_ROW_PAD   72                    // ushorts per LDS row (144 B)
#define TT_BUF_ELEMS (128 * TT_ROW_PAD)    // 9216 ushorts = 18432 B / buffer

__device__ __forceinline__ void tt_async_b128(unsigned ldsOff,
                                              unsigned long long gaddr) {
    asm volatile("global_load_async_to_lds_b128 %0, %1, off"
                 :: "v"(ldsOff), "v"(gaddr) : "memory");
}

__global__ __launch_bounds__(256) void tt_wmma_gemm(const unsigned short* __restrict__ Xb,
                                                    const unsigned short* __restrict__ Gt,
                                                    const float* __restrict__ bias,
                                                    float* __restrict__ out) {
    __shared__ unsigned short As[2 * TT_BUF_ELEMS];   // 36 KB
    __shared__ unsigned short Bs[2 * TT_BUF_ELEMS];   // 36 KB

    const int tid   = threadIdx.x;
    const int lane  = tid & 31;
    const int ln    = lane & 15;
    const int half  = lane >> 4;           // 0 or 1
    const int w     = tid >> 5;            // wave 0..7
    const int wr    = w >> 1;              // 0..3 -> M offset wr*32
    const int wc    = w & 1;               // 0..1 -> N offset wc*64
    const int nBase = blockIdx.x * 128;
    const int mBase = blockIdx.y * 128;

    // ---- async staging: 1024 16-byte chunks per panel per K-step, 4/thread.
    //      chunk c: row = c>>3, seg = c&7 (8-elem K segment).
    const unsigned asBase = (unsigned)(size_t)&As[0];
    const unsigned bsBase = (unsigned)(size_t)&Bs[0];

    unsigned long long gA[4], gB[4];
    unsigned lA[4], lB[4];
    #pragma unroll
    for (int i = 0; i < 4; ++i) {
        const int c   = tid + i * 256;
        const int row = c >> 3;
        const int seg = c & 7;
        gA[i] = (unsigned long long)(size_t)Xb +
                ((size_t)(mBase + row) * 4096 + seg * 8) * 2;
        gB[i] = (unsigned long long)(size_t)Gt +
                ((size_t)(nBase + row) * 4096 + seg * 8) * 2;
        lA[i] = asBase + row * (TT_ROW_PAD * 2) + seg * 16;
        lB[i] = bsBase + row * (TT_ROW_PAD * 2) + seg * 16;
    }

    float bv[4];
    #pragma unroll
    for (int ni = 0; ni < 4; ++ni)
        bv[ni] = bias[nBase + wc * 64 + ni * 16 + ln];

    v8f acc[8];
    const v8f vzero = {0.f, 0.f, 0.f, 0.f, 0.f, 0.f, 0.f, 0.f};
    #pragma unroll
    for (int i = 0; i < 8; ++i) acc[i] = vzero;

    // LDS fragment base offsets (elements) for this wave/lane.
    const int aRowOff = (wr * 32 + ln) * TT_ROW_PAD;      // + mi*16*TT_ROW_PAD
    const int bRowOff = (wc * 64 + ln) * TT_ROW_PAD;      // + ni*16*TT_ROW_PAD

    // ---- prologue: stage K-step 0 into buffer 0 (8 async ops per wave)
    #pragma unroll
    for (int i = 0; i < 4; ++i) { tt_async_b128(lA[i], gA[i]); }
    #pragma unroll
    for (int i = 0; i < 4; ++i) { tt_async_b128(lB[i], gB[i]); }

    #define TT_COMPUTE_STEP(ab, bb)                                            \
        do {                                                                   \
            _Pragma("unroll")                                                  \
            for (int s = 0; s < 2; ++s) {                                      \
                Frag16 A[2], B[4];                                             \
                _Pragma("unroll")                                              \
                for (int mi = 0; mi < 2; ++mi) {                               \
                    const unsigned short* ap =                                 \
                        (ab) + aRowOff + mi * 16 * TT_ROW_PAD + s * 32;        \
                    A[mi].u[0] = *(const uint4*)(ap + half * 8);               \
                    A[mi].u[1] = *(const uint4*)(ap + 16 + half * 8);          \
                }                                                              \
                _Pragma("unroll")                                              \
                for (int ni = 0; ni < 4; ++ni) {                               \
                    const unsigned short* bp =                                 \
                        (bb) + bRowOff + ni * 16 * TT_ROW_PAD + s * 32;        \
                    B[ni].u[0] = *(const uint4*)(bp + half * 8);               \
                    B[ni].u[1] = *(const uint4*)(bp + 16 + half * 8);          \
                }                                                              \
                _Pragma("unroll")                                              \
                for (int mi = 0; mi < 2; ++mi) {                               \
                    _Pragma("unroll")                                          \
                    for (int ni = 0; ni < 4; ++ni) {                           \
                        acc[mi * 4 + ni] =                                     \
                            __builtin_amdgcn_wmma_f32_16x16x32_bf16(           \
                                false, A[mi].v, false, B[ni].v,                \
                                (short)0, acc[mi * 4 + ni], false, false);     \
                    }                                                          \
                }                                                              \
            }                                                                  \
        } while (0)

    // ---- steady state: 63 branch-free iterations (last K-step peeled)
    int ibuf = 0;
    #pragma unroll 1
    for (int kk = 0; kk < 4096 - 64; kk += 64, ibuf ^= 1) {
        const unsigned nb  = (unsigned)(ibuf ^ 1) * (TT_BUF_ELEMS * 2);
        const unsigned gko = (unsigned)(kk + 64) * 2;      // bytes
        #pragma unroll
        for (int i = 0; i < 4; ++i) { tt_async_b128(lA[i] + nb, gA[i] + gko); }
        #pragma unroll
        for (int i = 0; i < 4; ++i) { tt_async_b128(lB[i] + nb, gB[i] + gko); }
        asm volatile("s_wait_asynccnt 0x8" ::: "memory");  // current buf ready
        __syncthreads();

        const unsigned short* ab = &As[ibuf * TT_BUF_ELEMS];
        const unsigned short* bb = &Bs[ibuf * TT_BUF_ELEMS];
        TT_COMPUTE_STEP(ab, bb);

        __syncthreads();   // reads done before this buffer is refilled
    }

    // ---- peeled final K-step
    {
        asm volatile("s_wait_asynccnt 0x0" ::: "memory");
        __syncthreads();
        const unsigned short* ab = &As[ibuf * TT_BUF_ELEMS];
        const unsigned short* bb = &Bs[ibuf * TT_BUF_ELEMS];
        TT_COMPUTE_STEP(ab, bb);
    }

    // ---- epilogue: C layout store + bias
    #pragma unroll
    for (int mi = 0; mi < 2; ++mi) {
        const int Mb = mBase + wr * 32 + mi * 16 + half * 8;
        #pragma unroll
        for (int ni = 0; ni < 4; ++ni) {
            const int N = nBase + wc * 64 + ni * 16 + ln;
            const v8f c = acc[mi * 4 + ni];
            #pragma unroll
            for (int v = 0; v < 8; ++v)
                out[(size_t)(Mb + v) * 4096 + N] = c[v] + bv[ni];
        }
    }
}

// ---------------------------------------------------------------------------
// Host launcher. Inputs: x, g0, g1, g2, bias (all fp32).
// Workspace: A1 fp32 @0 (16 MiB) | Gt bf16 @16MiB (32 MiB) | Xb bf16 @48MiB.
// ---------------------------------------------------------------------------
extern "C" void kernel_launch(void* const* d_in, const int* in_sizes, int n_in,
                              void* d_out, int out_size, void* d_ws, size_t ws_size,
                              hipStream_t stream) {
    const float* x    = (const float*)d_in[0];
    const float* g0   = (const float*)d_in[1];
    const float* g1   = (const float*)d_in[2];
    const float* g2   = (const float*)d_in[3];
    const float* bias = (const float*)d_in[4];
    float* out = (float*)d_out;

    char* ws = (char*)d_ws;
    float*          A1 = (float*)(ws);
    unsigned short* Gt = (unsigned short*)(ws + (size_t)(16 << 20));
    unsigned short* Xb = (unsigned short*)(ws + (size_t)(48 << 20));

    // 1) A1 = g0 @ g1 : 256 x 16384 fp32
    tt_build_a1<<<4096, 256, 0, stream>>>(g0, g1, A1);
    // 2) Gt = (A1v @ g2)^T : 4096 x 4096 bf16, N-major
    tt_build_gt<<<8192, 256, 0, stream>>>(A1, g2, Gt);
    // 3) Xb = bf16(X) : 8192 x 4096
    tt_cvt_x<<<16384, 256, 0, stream>>>(x, Xb);
    // 4) out = Xb @ G + bias via WMMA, 128x128 macro-tiles, async LDS staging
    tt_wmma_gemm<<<dim3(32, 64), 256, 0, stream>>>(Xb, Gt, bias, out);
}
